// SioConvNetBlock_33294586478862
// MI455X (gfx1250) — matrix operand order repaired and checked by reading, not compiled, not run on hardware
//
#include <hip/hip_runtime.h>
#include <hip/hip_bf16.h>

typedef __bf16 bf16;
typedef __attribute__((ext_vector_type(8)))  bf16  v8bf;
typedef __attribute__((ext_vector_type(16))) bf16  v16bf;
typedef __attribute__((ext_vector_type(8)))  float v8f;

#define DIM 512
#define DT 64
#define DFF 2048
#define BATCH 2
#define SEQ 512
#define TOK (BATCH * SEQ)   // 1024 rows

// ---------------- fp32 -> bf16 weight conversion ----------------
__global__ void cvt_f32_bf16(const float* __restrict__ src, bf16* __restrict__ dst, int n) {
  int i = blockIdx.x * blockDim.x + threadIdx.x;
  if (i < n) dst[i] = (bf16)src[i];
}

// ---------------- LayerNorm: one wave per 512-wide row ----------------
__global__ __launch_bounds__(128)
void layernorm_to_bf16(const float* __restrict__ x, const float* __restrict__ g,
                       const float* __restrict__ bta, bf16* __restrict__ out, int rows) {
  int wave = (blockIdx.x * blockDim.x + threadIdx.x) >> 5;
  int lane = threadIdx.x & 31;
  if (wave >= rows) return;
  const float* xr = x + (size_t)wave * DIM;
  float v[16];
  float s = 0.f;
#pragma unroll
  for (int j = 0; j < 16; ++j) { v[j] = xr[j * 32 + lane]; s += v[j]; }
#pragma unroll
  for (int off = 16; off >= 1; off >>= 1) s += __shfl_xor(s, off, 32);
  float mu = s * (1.0f / DIM);
  float q = 0.f;
#pragma unroll
  for (int j = 0; j < 16; ++j) { float d = v[j] - mu; q += d * d; }
#pragma unroll
  for (int off = 16; off >= 1; off >>= 1) q += __shfl_xor(q, off, 32);
  float rstd = rsqrtf(q * (1.0f / DIM) + 1e-5f);
  bf16* orow = out + (size_t)wave * DIM;
#pragma unroll
  for (int j = 0; j < 16; ++j) {
    int c = j * 32 + lane;
    orow[c] = (bf16)((v[j] - mu) * rstd * g[c] + bta[c]);
  }
}

// ---------------- WMMA fragment loaders (bf16, 16x16x32) ----------------
// A (16x32, MxK): lane half selects two 8-wide K runs at k0+half*8 and k0+16+half*8
__device__ __forceinline__ v16bf ld_frag_a(const bf16* p) {
  union { v16bf v; v8bf h[2]; } u;
  u.h[0] = *(const v8bf*)(p);
  u.h[1] = *(const v8bf*)(p + 16);
  return u.v;
}
// B (32x16, KxN) built from row-major W(N,K): lane half selects two 8-wide K runs at k0+half*8 and k0+16+half*8
__device__ __forceinline__ v16bf ld_frag_b(const bf16* p) {
  union { v16bf v; v8bf h[2]; } u;
  u.h[0] = *(const v8bf*)(p);
  u.h[1] = *(const v8bf*)(p + 16);
  return u.v;
}

__device__ __forceinline__ float silu_f(float t) { return t / (1.f + __expf(-t)); }

// ---------------- Generic WMMA GEMM: C = A(M,K) @ W(N,K)^T + epilogue ----------------
// MODE 0: outf = acc                                  (raw f32)
// MODE 1: outb = bf16(silu(acc + bias0[n]))           (FFN1)
// MODE 2: split N=128: n<64 -> aux_f = silu(acc+bias0[n])  (gate y)
//                      n>=64 -> outf/aux_b = acc+bias1[n-64] (u f32 + bf16)
// MODE 3: outf = acc + bias0[n] + res[m*N+n]          (proj + residual)
template <int MODE>
__global__ __launch_bounds__(128)
void gemm_wmma_bf16(const bf16* __restrict__ A, const bf16* __restrict__ W,
                    int M, int N, int K,
                    const float* __restrict__ bias0, const float* __restrict__ bias1,
                    const float* __restrict__ res,
                    float* __restrict__ outf, bf16* __restrict__ outb,
                    float* __restrict__ aux_f, bf16* __restrict__ aux_b) {
  const int lane = threadIdx.x & 31;
  const int wave = threadIdx.x >> 5;
  const int half = lane >> 4;
  const int lr   = lane & 15;
  const int m0 = blockIdx.y * 128 + (wave >> 1) * 64;
  const int n0 = blockIdx.x * 128 + (wave & 1) * 64;

  v8f zero = {};
  v8f acc[4][4];
#pragma unroll
  for (int i = 0; i < 4; ++i)
#pragma unroll
    for (int j = 0; j < 4; ++j) acc[i][j] = zero;

  for (int k0 = 0; k0 < K; k0 += 32) {
    v16bf af[4], wf[4];
#pragma unroll
    for (int i = 0; i < 4; ++i)
      af[i] = ld_frag_a(A + (size_t)(m0 + i * 16 + lr) * K + k0 + half * 8);
#pragma unroll
    for (int j = 0; j < 4; ++j)
      wf[j] = ld_frag_b(W + (size_t)(n0 + j * 16 + lr) * K + k0 + half * 8);
#pragma unroll
    for (int i = 0; i < 4; ++i)
#pragma unroll
      for (int j = 0; j < 4; ++j)
        acc[i][j] = __builtin_amdgcn_wmma_f32_16x16x32_bf16(
            false, af[i], false, wf[j], (short)0, acc[i][j], false, false);
  }

#pragma unroll
  for (int i = 0; i < 4; ++i) {
#pragma unroll
    for (int j = 0; j < 4; ++j) {
#pragma unroll
      for (int r = 0; r < 8; ++r) {
        int m = m0 + i * 16 + r + half * 8;
        int n = n0 + j * 16 + lr;
        float v = acc[i][j][r];
        if (MODE == 0) {
          outf[(size_t)m * N + n] = v;
        } else if (MODE == 1) {
          outb[(size_t)m * N + n] = (bf16)silu_f(v + bias0[n]);
        } else if (MODE == 2) {
          if (n < 64) {
            aux_f[(size_t)m * 64 + n] = silu_f(v + bias0[n]);
          } else {
            float t = v + bias1[n - 64];
            outf[(size_t)m * 64 + (n - 64)] = t;
            aux_b[(size_t)m * 64 + (n - 64)] = (bf16)t;
          }
        } else {  // MODE == 3
          size_t idx = (size_t)m * N + n;
          outf[idx] = v + bias0[n] + res[idx];
        }
      }
    }
  }
}

// ---------------- SioConv parallel reverse scan ----------------
// Reference: h_i = sum_{k>=i-1} prod_{j=i..k} a_j * xin_k, xin = [hidden, u_0..u_{L-2}],
//            h_{L-1} += u_{L-1}.  Linear form: r_i = a_i*(xin_i + r_{i+1}), r_L = 0,
//            h_i = xin_{i-1} + r_i (h_0 = r_0).
// Parallelization: one wave per (b,d) channel; lane l owns chunk [16l, 16l+16).
// Pass 1 builds the chunk's affine map r_left = P*r_right + Q (complex);
// a 5-step shuffle suffix-scan composes maps across lanes; pass 2 applies the
// entering value, fuses the a-normalization and h*y -> bf16.
__global__ __launch_bounds__(128)
void sioconv_scan(const float* __restrict__ aout, const float* __restrict__ u,
                  const float* __restrict__ y,
                  const float* __restrict__ hid_re, const float* __restrict__ hid_im,
                  bf16* __restrict__ hy) {
  const int lane = threadIdx.x & 31;
  const int ch = blockIdx.x * 4 + (threadIdx.x >> 5);  // 0..127
  const int b = ch >> 6, d = ch & 63;
  const int base = lane * 16;
  const size_t rowb = (size_t)b * SEQ;

  float ar[16], ai[16], xr16[16];
  float Pr = 1.f, Pi = 0.f, Qr = 0.f, Qi = 0.f;

  // Pass 1: normalize a, build chunk affine map (iterate chunk right-to-left)
#pragma unroll
  for (int t = 15; t >= 0; --t) {
    int i = base + t;
    const float* ap = aout + (rowb + i) * 128 + 2 * d;
    float re = ap[0], im = ap[1];
    float s2 = re * re + im * im;
    float sc = rsqrtf(s2) * __expf(-s2);
    float are = re * sc, aim = im * sc;
    ar[t] = are; ai[t] = aim;
    float xr, xi;
    if (i == 0) { xr = hid_re[d]; xi = hid_im[d]; }
    else        { xr = u[(rowb + i - 1) * 64 + d]; xi = 0.f; }
    xr16[t] = xr;
    // Q <- a*(x + Q), P <- a*P
    float tr = xr + Qr, ti = xi + Qi;
    Qr = are * tr - aim * ti;
    Qi = are * ti + aim * tr;
    float pr = are * Pr - aim * Pi;
    float pi = are * Pi + aim * Pr;
    Pr = pr; Pi = pi;
  }

  // Suffix inclusive scan of affine maps across lanes (self left-composed with lane+off)
#pragma unroll
  for (int off = 1; off < 32; off <<= 1) {
    float Pr2 = __shfl_down(Pr, off, 32);
    float Pi2 = __shfl_down(Pi, off, 32);
    float Qr2 = __shfl_down(Qr, off, 32);
    float Qi2 = __shfl_down(Qi, off, 32);
    if (lane + off < 32) {
      float nQr = Pr * Qr2 - Pi * Qi2 + Qr;
      float nQi = Pr * Qi2 + Pi * Qr2 + Qi;
      float nPr = Pr * Pr2 - Pi * Pi2;
      float nPi = Pr * Pi2 + Pi * Pr2;
      Qr = nQr; Qi = nQi; Pr = nPr; Pi = nPi;
    }
  }
  // entering value for this chunk = Q of lane+1's inclusive suffix (0 for lane 31)
  float r_re = __shfl_down(Qr, 1, 32);
  float r_im = __shfl_down(Qi, 1, 32);
  if (lane == 31) { r_re = 0.f; r_im = 0.f; }

  // Pass 2: apply recurrence within chunk, emit h*y
  float hre0 = hid_re[d];
#pragma unroll
  for (int t = 15; t >= 0; --t) {
    int i = base + t;
    float xi = (i == 0) ? hid_im[d] : 0.f;
    float tr = xr16[t] + r_re, ti = xi + r_im;
    float nr = ar[t] * tr - ai[t] * ti;
    float ni = ar[t] * ti + ai[t] * tr;
    r_re = nr; r_im = ni;
    float h_re;
    if (i == 0)      h_re = r_re;
    else if (i == 1) h_re = hre0 + r_re;
    else             h_re = u[(rowb + i - 2) * 64 + d] + r_re;
    if (i == SEQ - 1) h_re += u[(rowb + i) * 64 + d];
    hy[(rowb + i) * 64 + d] = (bf16)(h_re * y[(rowb + i) * 64 + d]);
  }
}

// ---------------- launcher ----------------
extern "C" void kernel_launch(void* const* d_in, const int* in_sizes, int n_in,
                              void* d_out, int out_size, void* d_ws, size_t ws_size,
                              hipStream_t stream) {
  const float* x        = (const float*)d_in[0];
  const float* ln1_g    = (const float*)d_in[1];
  const float* ln1_b    = (const float*)d_in[2];
  const float* fc_w     = (const float*)d_in[3];
  const float* fc_b     = (const float*)d_in[4];
  const float* lin_in_w = (const float*)d_in[5];
  const float* lin_in_b = (const float*)d_in[6];
  const float* wa       = (const float*)d_in[7];
  const float* hid_re   = (const float*)d_in[8];
  const float* hid_im   = (const float*)d_in[9];
  const float* lo_w     = (const float*)d_in[10];
  const float* lo_b     = (const float*)d_in[11];
  const float* ln2_g    = (const float*)d_in[12];
  const float* ln2_b    = (const float*)d_in[13];
  const float* w1       = (const float*)d_in[14];
  const float* b1       = (const float*)d_in[15];
  const float* w2       = (const float*)d_in[16];
  const float* b2       = (const float*)d_in[17];
  float* out = (float*)d_out;

  // workspace carve-up (256B aligned regions)
  char* ws = (char*)d_ws;
  size_t off = 0;
  auto alloc = [&](size_t bytes) { char* p = ws + off; off = (off + bytes + 255) & ~(size_t)255; return p; };
  bf16*  xn_bf  = (bf16*) alloc((size_t)TOK * DIM * 2);
  bf16*  wcat   = (bf16*) alloc((size_t)128 * DIM * 2);     // [fc_w ; lin_in_w]
  bf16*  wa_bf  = (bf16*) alloc((size_t)128 * DT * 2);
  bf16*  lw_bf  = (bf16*) alloc((size_t)DIM * DT * 2);
  bf16*  w1_bf  = (bf16*) alloc((size_t)DFF * DIM * 2);
  bf16*  w2_bf  = (bf16*) alloc((size_t)DIM * DFF * 2);
  float* y_f    = (float*)alloc((size_t)TOK * DT * 4);
  float* u_f    = (float*)alloc((size_t)TOK * DT * 4);
  bf16*  u_bf   = (bf16*) alloc((size_t)TOK * DT * 2);
  float* aoutf  = (float*)alloc((size_t)TOK * 128 * 4);
  bf16*  hy_bf  = (bf16*) alloc((size_t)TOK * DT * 2);
  float* x2     = (float*)alloc((size_t)TOK * DIM * 4);
  bf16*  xn2_bf = (bf16*) alloc((size_t)TOK * DIM * 2);
  bf16*  t_bf   = (bf16*) alloc((size_t)TOK * DFF * 2);
  (void)ws_size; (void)in_sizes; (void)n_in; (void)out_size;

  auto cvt = [&](const float* s, bf16* dst, int n) {
    cvt_f32_bf16<<<(n + 255) / 256, 256, 0, stream>>>(s, dst, n);
  };
  cvt(fc_w,     wcat,            64 * DIM);
  cvt(lin_in_w, wcat + 64 * DIM, 64 * DIM);
  cvt(wa,       wa_bf,           128 * DT);
  cvt(lo_w,     lw_bf,           DIM * DT);
  cvt(w1,       w1_bf,           DFF * DIM);
  cvt(w2,       w2_bf,           DIM * DFF);

  // LN1: x -> xn (bf16)
  layernorm_to_bf16<<<TOK / 4, 128, 0, stream>>>(x, ln1_g, ln1_b, xn_bf, TOK);

  // gate + input proj: (1024x512)@(128x512)^T -> y (silu, f32), u (f32+bf16)
  gemm_wmma_bf16<2><<<dim3(1, TOK / 128), 128, 0, stream>>>(
      xn_bf, wcat, TOK, 128, DIM, fc_b, lin_in_b, nullptr, u_f, nullptr, y_f, u_bf);

  // a = u @ wa^T : (1024x64)@(128x64)^T -> raw complex pairs (f32)
  gemm_wmma_bf16<0><<<dim3(1, TOK / 128), 128, 0, stream>>>(
      u_bf, wa_bf, TOK, 128, DT, nullptr, nullptr, nullptr, aoutf, nullptr, nullptr, nullptr);

  // normalize-a + parallel reverse recurrence + h*y -> bf16 (one wave per channel)
  sioconv_scan<<<32, 128, 0, stream>>>(aoutf, u_f, y_f, hid_re, hid_im, hy_bf);

  // x2 = (h*y) @ lin_out_w^T + lin_out_b + x
  gemm_wmma_bf16<3><<<dim3(DIM / 128, TOK / 128), 128, 0, stream>>>(
      hy_bf, lw_bf, TOK, DIM, DT, lo_b, nullptr, x, x2, nullptr, nullptr, nullptr);

  // LN2: x2 -> xn2 (bf16)
  layernorm_to_bf16<<<TOK / 4, 128, 0, stream>>>(x2, ln2_g, ln2_b, xn2_bf, TOK);

  // FFN1: t = silu(xn2 @ w1^T + b1) (bf16)
  gemm_wmma_bf16<1><<<dim3(DFF / 128, TOK / 128), 128, 0, stream>>>(
      xn2_bf, w1_bf, TOK, DFF, DIM, b1, nullptr, nullptr, nullptr, t_bf, nullptr, nullptr);

  // FFN2: out = t @ w2^T + b2 + x2
  gemm_wmma_bf16<3><<<dim3(DIM / 128, TOK / 128), 128, 0, stream>>>(
      t_bf, w2_bf, TOK, DIM, DFF, b2, nullptr, x2, out, nullptr, nullptr, nullptr);
}
